// TransformerEncoderLayerRPE_45174466019881
// MI455X (gfx1250) — matrix operand order repaired and checked by reading, compiled, not run on hardware
//
#include <hip/hip_runtime.h>

typedef __bf16 bf16_t;
typedef __attribute__((ext_vector_type(16))) __bf16 v16bf;
typedef __attribute__((ext_vector_type(8)))  float   v8f;
typedef __attribute__((ext_vector_type(4)))  int     v4i;
typedef __attribute__((ext_vector_type(8)))  int     v8i;
typedef __attribute__((ext_vector_type(4)))  unsigned int v4u;

#define D_MODEL 1024
#define NHEAD   16
#define DHEAD   64
#define DFF     4096
#define SEQ     2048
#define BATCH   2
#define ROWS    (BATCH * SEQ)   // 4096
#define MAXDIST 32

#define TM 128
#define TN 128
#define TK 64
#define LDA_SH (TK + 8)   // padded K stride (elems) for A tile rows
#define LDT_SH (TK + 8)   // padded K stride (elems) for BT tile rows

#ifndef __has_builtin
#define __has_builtin(x) 0
#endif
#if __has_builtin(__builtin_amdgcn_global_load_async_to_lds_b128)
#define HAS_ASYNC_LDS 1
#else
#define HAS_ASYNC_LDS 0
#endif
#if __has_builtin(__builtin_amdgcn_tensor_load_to_lds)
#define HAS_TDM 1
#else
#define HAS_TDM 0
#endif

// ---------------------------------------------------------------------------
// 16-byte global->LDS copy: async (ASYNCcnt-tracked) when available
// ---------------------------------------------------------------------------
static __device__ __forceinline__ void cp16(const bf16_t* g, bf16_t* l) {
#if HAS_ASYNC_LDS
  __builtin_amdgcn_global_load_async_to_lds_b128(
      (__attribute__((address_space(1))) v4i*)g,
      (__attribute__((address_space(3))) v4i*)l, 0, 0);
#else
  *(int4*)l = *(const int4*)g;
#endif
}
static __device__ __forceinline__ void cp_wait() {
#if HAS_ASYNC_LDS
#if __has_builtin(__builtin_amdgcn_s_wait_asynccnt)
  __builtin_amdgcn_s_wait_asynccnt(0);
#else
  asm volatile("s_wait_asynccnt 0" ::: "memory");
#endif
#endif
}

// ---------------------------------------------------------------------------
// Tensor Data Mover: 2D tile (rows x tileDW dwords) from global to LDS with
// hardware row padding. D# built per CDNA5 ISA 8.3-8.6. dword data (size=4B).
// Toolchain uses the 6-arg builtin: (g0, g1, g2, g3, extra_v8i, cpol).
// ---------------------------------------------------------------------------
static __device__ __forceinline__ unsigned lds_off(const void* p) {
  return (unsigned)(unsigned long long)(__attribute__((address_space(3))) const char*)p;
}
#if HAS_TDM
static __device__ __forceinline__ void tdm_load_2d(const void* gsrc, unsigned ldsOff,
                                                   unsigned strideDW, unsigned tileDW,
                                                   unsigned rows, unsigned padI,
                                                   unsigned padA) {
  const unsigned long long ga = (unsigned long long)gsrc;
  v4u g0 = {1u,                                   // count=1, user descriptor
            ldsOff,                               // lds_addr (bytes)
            (unsigned)ga,                         // global_addr[31:0]
            (unsigned)((ga >> 32) & 0x1ffffffu) | (2u << 30)};  // addr hi | type=2
  v8i g1;
  g1[0] = (int)((2u << 16) | (1u << 20) | (padI << 22) | (padA << 25));
  g1[1] = (int)((strideDW & 0xffffu) << 16);                 // tensor_dim0 lo16
  g1[2] = (int)(((strideDW >> 16) & 0xffffu) | ((rows & 0xffffu) << 16));
  g1[3] = (int)(((rows >> 16) & 0xffffu) | (tileDW << 16));  // dim1 hi | tile_dim0
  g1[4] = (int)rows;                                          // tile_dim1, tile_dim2=0
  g1[5] = (int)strideDW;                                      // tensor_dim0_stride lo
  g1[6] = 0;
  g1[7] = 0;
  v4i z4 = {0, 0, 0, 0};
  v8i z8 = {0, 0, 0, 0, 0, 0, 0, 0};
  __builtin_amdgcn_tensor_load_to_lds(g0, g1, z4, z4, z8, 0);
}
static __device__ __forceinline__ void tdm_wait() {
#if __has_builtin(__builtin_amdgcn_s_wait_tensorcnt)
  __builtin_amdgcn_s_wait_tensorcnt(0);
#else
  asm volatile("s_wait_tensorcnt 0" ::: "memory");
#endif
}
#endif

// ---------------------------------------------------------------------------
// WMMA helpers (CDNA5 v_wmma_f32_16x16x32_bf16, wave32)
// ---------------------------------------------------------------------------
static __device__ __forceinline__ v8f wmma_bf16(v16bf a, v16bf b, v8f c) {
  return __builtin_amdgcn_wmma_f32_16x16x32_bf16(false, a, false, b, (short)0, c,
                                                 false, false);
}

// A fragment, 16x32 bf16, row-major source, stride ld.
static __device__ __forceinline__ v16bf load_frag_a(const bf16_t* p, int ld) {
  const int lane = threadIdx.x & 31;
  const bf16_t* r = p + (lane & 15) * ld + (lane >> 4) * 16;
  v16bf f;
#pragma unroll
  for (int i = 0; i < 8; ++i) { f[i] = r[i]; f[i + 8] = r[8 + i]; }
  return f;
}

// B fragment, 32x16, from TRANSPOSED storage BT[n][k] (stride ld).
static __device__ __forceinline__ v16bf load_frag_b_col(const bf16_t* pT, int ld) {
  const int lane = threadIdx.x & 31;
  const bf16_t* r = pT + (lane & 15) * ld + (lane >> 4) * 16;
  v16bf f;
#pragma unroll
  for (int j = 0; j < 16; ++j) f[j] = r[j];
  return f;
}

// half-wave (16-lane) butterfly reductions. DPP path: xor masks {1,2,4,15}
// generate the full 16-lane group, all in the VALU (no LDS bpermute).
#if __has_builtin(__builtin_amdgcn_update_dpp)
#define DPPF(v, ctrl)                                                         \
  __int_as_float(__builtin_amdgcn_update_dpp(0, __float_as_int(v), ctrl, 0xf, \
                                             0xf, true))
static __device__ __forceinline__ float half_max(float v) {
  v = fmaxf(v, DPPF(v, 0xB1));   // quad_perm [1,0,3,2]  (xor1)
  v = fmaxf(v, DPPF(v, 0x4E));   // quad_perm [2,3,0,1]  (xor2)
  v = fmaxf(v, DPPF(v, 0x141));  // row_half_mirror      (xor4)
  v = fmaxf(v, DPPF(v, 0x140));  // row_mirror           (xor15)
  return v;
}
static __device__ __forceinline__ float half_sum(float v) {
  v += DPPF(v, 0xB1);
  v += DPPF(v, 0x4E);
  v += DPPF(v, 0x141);
  v += DPPF(v, 0x140);
  return v;
}
#else
static __device__ __forceinline__ float half_max(float v) {
#pragma unroll
  for (int m = 1; m < 16; m <<= 1) v = fmaxf(v, __shfl_xor(v, m, 32));
  return v;
}
static __device__ __forceinline__ float half_sum(float v) {
#pragma unroll
  for (int m = 1; m < 16; m <<= 1) v += __shfl_xor(v, m, 32);
  return v;
}
#endif

// ---------------------------------------------------------------------------
// fp32 -> bf16 (plain)
// ---------------------------------------------------------------------------
__global__ void k_cvt(const float* __restrict__ s, bf16_t* __restrict__ d, int n) {
  const int i = blockIdx.x * blockDim.x + threadIdx.x;
  if (i < n) d[i] = (bf16_t)s[i];
}

// fp32 [K][N] -> bf16 transposed [N][K], LDS-tiled
__global__ __launch_bounds__(256) void k_cvt_t(const float* __restrict__ s,
                                               bf16_t* __restrict__ d,
                                               int Kdim, int Ndim) {
  __shared__ float tile[32][33];
  const int k0 = blockIdx.y * 32;
  const int n0 = blockIdx.x * 32;
  const int tx = threadIdx.x & 31;
  const int ty = threadIdx.x >> 5;
#pragma unroll
  for (int i = 0; i < 4; ++i)
    tile[ty + i * 8][tx] = s[(size_t)(k0 + ty + i * 8) * Ndim + n0 + tx];
  __syncthreads();
#pragma unroll
  for (int i = 0; i < 4; ++i)
    d[(size_t)(n0 + ty + i * 8) * Kdim + k0 + tx] = (bf16_t)tile[tx][ty + i * 8];
}

// ---------------------------------------------------------------------------
// Tiled bf16 WMMA GEMM: C[MxN] = A[MxK] * BT^T + bias, BT is [N][K].
// Double-buffered LDS; tile fills via TDM (TENSORcnt) when available, else
// async-LDS. K-step 64 -> 16 WMMAs per wave per barrier.
// ---------------------------------------------------------------------------
template <int MODE>
__global__ __launch_bounds__(256) void k_gemm(const bf16_t* __restrict__ A,
                                              const bf16_t* __restrict__ BT,
                                              const float* __restrict__ bias,
                                              float* __restrict__ outF,
                                              bf16_t* __restrict__ outB,
                                              int M, int N, int K) {
  __shared__ bf16_t a_sh[2][TM][LDA_SH];
  __shared__ bf16_t bt_sh[2][TN][LDT_SH];

  const int t  = threadIdx.x;
  const int m0 = blockIdx.y * TM;
  const int n0 = blockIdx.x * TN;
  const int wv = t >> 5;
  const int wm = wv >> 1;                 // 0..3 (32-row strip)
  const int wn = wv & 1;                  // 0..1 (64-col strip)
  const int lane = t & 31, cn = lane & 15, gm = lane >> 4;

  v8f acc[2][4] = {};

#if !HAS_TDM
  const int ar = t >> 1, ac = (t & 1) * 32;   // fills: 128 rows x 64 elems
#endif

  auto fill = [&](int buf, int kk) {
#if HAS_TDM
    if (wv == 0) {
      // rows x 32-dword tile, pad +4 dwords every 32 dwords (matches +8 bf16)
      tdm_load_2d(A + (size_t)m0 * K + kk, lds_off(&a_sh[buf][0][0]),
                  (unsigned)(K >> 1), TK >> 1, TM, 4, 3);
      tdm_load_2d(BT + (size_t)n0 * K + kk, lds_off(&bt_sh[buf][0][0]),
                  (unsigned)(K >> 1), TK >> 1, TN, 4, 3);
    }
#else
    const bf16_t* asrc = A + (size_t)(m0 + ar) * K + kk + ac;
    cp16(asrc, &a_sh[buf][ar][ac]);      cp16(asrc + 8, &a_sh[buf][ar][ac + 8]);
    cp16(asrc + 16, &a_sh[buf][ar][ac + 16]); cp16(asrc + 24, &a_sh[buf][ar][ac + 24]);
    const bf16_t* bsrc = BT + (size_t)(n0 + ar) * K + kk + ac;
    cp16(bsrc, &bt_sh[buf][ar][ac]);     cp16(bsrc + 8, &bt_sh[buf][ar][ac + 8]);
    cp16(bsrc + 16, &bt_sh[buf][ar][ac + 16]); cp16(bsrc + 24, &bt_sh[buf][ar][ac + 24]);
#endif
  };
  auto wait_fill = [&]() {
#if HAS_TDM
    if (wv == 0) tdm_wait();
#else
    cp_wait();
#endif
  };

  fill(0, 0);
  wait_fill();
  __syncthreads();

  int buf = 0;
  for (int k0 = 0; k0 < K; k0 += TK, buf ^= 1) {
    if (k0 + TK < K) fill(buf ^ 1, k0 + TK);   // DMA prefetch of next tile

#pragma unroll
    for (int ks = 0; ks < TK; ks += 32) {
      v16bf fa[2], fb[4];
#pragma unroll
      for (int i = 0; i < 2; ++i)
        fa[i] = load_frag_a(&a_sh[buf][wm * 32 + i * 16][ks], LDA_SH);
#pragma unroll
      for (int j = 0; j < 4; ++j)
        fb[j] = load_frag_b_col(&bt_sh[buf][wn * 64 + j * 16][ks], LDT_SH);
#pragma unroll
      for (int i = 0; i < 2; ++i)
#pragma unroll
        for (int j = 0; j < 4; ++j) acc[i][j] = wmma_bf16(fa[i], fb[j], acc[i][j]);
    }

    wait_fill();
    __syncthreads();
  }

  // Epilogue. C layout: n = lane&15, m = vgpr + 8*(lane>>4).
#pragma unroll
  for (int i = 0; i < 2; ++i) {
#pragma unroll
    for (int j = 0; j < 4; ++j) {
      const int col = n0 + wn * 64 + j * 16 + cn;
      const float bv = bias[col];
#pragma unroll
      for (int r = 0; r < 8; ++r) {
        const int row = m0 + wm * 32 + i * 16 + r + 8 * gm;
        const float v = acc[i][j][r] + bv;
        if (MODE == 0) {
          outF[(size_t)row * N + col] = v;
        } else if (MODE == 1) {
          outB[(size_t)row * N + col] = (bf16_t)fmaxf(v, 0.0f);
        } else if (MODE == 2) {
          const size_t idx =
              (((size_t)(row >> 11) * NHEAD + (col >> 6)) * SEQ + (row & (SEQ - 1))) *
                  DHEAD + (col & (DHEAD - 1));
          outB[idx] = (bf16_t)v;
        } else {  // MODE 3: V transposed [B,H,DHEAD,SEQ]
          const size_t idx =
              (((size_t)(row >> 11) * NHEAD + (col >> 6)) * DHEAD + (col & (DHEAD - 1))) *
                  SEQ + (row & (SEQ - 1));
          outB[idx] = (bf16_t)v;
        }
      }
    }
  }
}

// ---------------------------------------------------------------------------
// Flash attention with relative-position bias (async-LDS tile fills).
// Grid (SEQ/64, BATCH*NHEAD), 128 thr (4 waves), wave owns 16 Q rows.
// ---------------------------------------------------------------------------
__global__ __launch_bounds__(128) void k_attn(const bf16_t* __restrict__ qB,
                                              const bf16_t* __restrict__ kB,
                                              const bf16_t* __restrict__ vB,
                                              const float* __restrict__ rpe,
                                              bf16_t* __restrict__ ctxB) {
  __shared__ bf16_t k_sh[64 * 64];    // [key][dh]
  __shared__ bf16_t vT_sh[64 * 64];   // [dh][key]
  __shared__ bf16_t p_sh[4][16 * 64];
  __shared__ float rpe_sh[(2 * MAXDIST + 1) * NHEAD];

  const int t  = threadIdx.x;
  const int bh = blockIdx.y;
  const int h  = bh & (NHEAD - 1);
  const int b  = bh >> 4;
  const int q0 = blockIdx.x * 64;
  const int wv = t >> 5;
  const int lane = t & 31, cn = lane & 15, gm = lane >> 4;
  const float scale = 0.125f;   // 1/sqrt(64)

  for (int i = t; i < (2 * MAXDIST + 1) * NHEAD; i += 128) rpe_sh[i] = rpe[i];

  const bf16_t* qg = qB + ((size_t)bh * SEQ + q0 + wv * 16) * DHEAD;
  const v16bf aq0 = load_frag_a(qg, DHEAD);
  const v16bf aq1 = load_frag_a(qg + 32, DHEAD);

  float m_i[8], l_i[8];
  v8f o[4] = {};
#pragma unroll
  for (int r = 0; r < 8; ++r) { m_i[r] = -1e30f; l_i[r] = 0.0f; }

  const int lr = t >> 1, lc = (t & 1) * 32;
  const bf16_t* kgb  = kB + (size_t)bh * SEQ * DHEAD;
  const bf16_t* vgbT = vB + (size_t)bh * DHEAD * SEQ;

  __syncthreads();   // rpe_sh ready
  const float bias_neg = rpe_sh[h];
  const float bias_pos = rpe_sh[2 * MAXDIST * NHEAD + h];

  for (int k0 = 0; k0 < SEQ; k0 += 64) {
    {
      const bf16_t* ks = kgb + (size_t)(k0 + lr) * DHEAD + lc;
      bf16_t* kd = &k_sh[lr * 64 + lc];
      cp16(ks, kd); cp16(ks + 8, kd + 8);
      cp16(ks + 16, kd + 16); cp16(ks + 24, kd + 24);
      const bf16_t* vs = vgbT + (size_t)lr * SEQ + k0 + lc;
      bf16_t* vd = &vT_sh[lr * 64 + lc];
      cp16(vs, vd); cp16(vs + 8, vd + 8);
      cp16(vs + 16, vd + 16); cp16(vs + 24, vd + 24);
    }
    cp_wait();
    __syncthreads();

    v8f s[4] = {};
#pragma unroll
    for (int nn = 0; nn < 4; ++nn) {
      const v16bf b0 = load_frag_b_col(&k_sh[(nn * 16) * 64], 64);
      const v16bf b1 = load_frag_b_col(&k_sh[(nn * 16) * 64 + 32], 64);
      s[nn] = wmma_bf16(aq0, b0, s[nn]);
      s[nn] = wmma_bf16(aq1, b1, s[nn]);
    }

    const int sat = (k0 + 95 <= q0) ? -1 : ((k0 >= q0 + 95) ? 1 : 0);

#pragma unroll
    for (int r = 0; r < 8; ++r) {
      const int qrow = q0 + wv * 16 + r + 8 * gm;
      float rowm = -1e30f;
#pragma unroll
      for (int nn = 0; nn < 4; ++nn) {
        float bvb;
        if (sat < 0) bvb = bias_neg;
        else if (sat > 0) bvb = bias_pos;
        else {
          const int key = k0 + nn * 16 + cn;
          int rel = key - qrow;
          rel = rel < -MAXDIST ? -MAXDIST : (rel > MAXDIST ? MAXDIST : rel);
          bvb = rpe_sh[(rel + MAXDIST) * NHEAD + h];
        }
        const float val = s[nn][r] * scale + bvb;
        s[nn][r] = val;
        rowm = fmaxf(rowm, val);
      }
      rowm = half_max(rowm);
      const float mnew  = fmaxf(m_i[r], rowm);
      const float alpha = __expf(m_i[r] - mnew);
      float lsum = 0.0f;
#pragma unroll
      for (int nn = 0; nn < 4; ++nn) {
        const float p = __expf(s[nn][r] - mnew);
        lsum += p;
        p_sh[wv][(r + 8 * gm) * 64 + nn * 16 + cn] = (bf16_t)p;
      }
      lsum = half_sum(lsum);
      l_i[r] = l_i[r] * alpha + lsum;
      m_i[r] = mnew;
#pragma unroll
      for (int nn = 0; nn < 4; ++nn) o[nn][r] *= alpha;
    }

#pragma unroll
    for (int kk = 0; kk < 2; ++kk) {
      const v16bf ap = load_frag_a(&p_sh[wv][kk * 32], 64);
#pragma unroll
      for (int nn = 0; nn < 4; ++nn) {
        const v16bf bvf = load_frag_b_col(&vT_sh[(nn * 16) * 64 + kk * 32], 64);
        o[nn] = wmma_bf16(ap, bvf, o[nn]);
      }
    }
    __syncthreads();
  }

#pragma unroll
  for (int nn = 0; nn < 4; ++nn)
#pragma unroll
    for (int r = 0; r < 8; ++r) {
      const int l = q0 + wv * 16 + r + 8 * gm;
      const float val = o[nn][r] / l_i[r];
      ctxB[((size_t)b * SEQ + l) * D_MODEL + h * DHEAD + nn * 16 + cn] = (bf16_t)val;
    }
}

// ---------------------------------------------------------------------------
// Fused residual add + LayerNorm (one block per row, D=1024)
// ---------------------------------------------------------------------------
__global__ __launch_bounds__(256) void k_add_ln(const float* __restrict__ a,
                                                const float* __restrict__ bsrc,
                                                const float* __restrict__ g,
                                                const float* __restrict__ be,
                                                float* __restrict__ outF,
                                                bf16_t* __restrict__ outB) {
  __shared__ float red[8];
  const int row = blockIdx.x;
  const int t = threadIdx.x;
  const size_t base = (size_t)row * D_MODEL;

  float v[4];
  float sum = 0.0f;
#pragma unroll
  for (int i = 0; i < 4; ++i) {
    const int idx = t + i * 256;
    v[i] = a[base + idx] + bsrc[base + idx];
    sum += v[i];
  }
#pragma unroll
  for (int m = 1; m < 32; m <<= 1) sum += __shfl_xor(sum, m, 32);
  if ((t & 31) == 0) red[t >> 5] = sum;
  __syncthreads();
  float tot = 0.0f;
#pragma unroll
  for (int i = 0; i < 8; ++i) tot += red[i];
  const float mu = tot * (1.0f / D_MODEL);
  __syncthreads();

  float vs = 0.0f;
#pragma unroll
  for (int i = 0; i < 4; ++i) { const float d = v[i] - mu; vs += d * d; }
#pragma unroll
  for (int m = 1; m < 32; m <<= 1) vs += __shfl_xor(vs, m, 32);
  if ((t & 31) == 0) red[t >> 5] = vs;
  __syncthreads();
  float vtot = 0.0f;
#pragma unroll
  for (int i = 0; i < 8; ++i) vtot += red[i];
  const float rstd = rsqrtf(vtot * (1.0f / D_MODEL) + 1e-5f);

#pragma unroll
  for (int i = 0; i < 4; ++i) {
    const int idx = t + i * 256;
    const float o = (v[i] - mu) * rstd * g[idx] + be[idx];
    outF[base + idx] = o;
    if (outB) outB[base + idx] = (bf16_t)o;
  }
}

// ---------------------------------------------------------------------------
// Launch
// ---------------------------------------------------------------------------
extern "C" void kernel_launch(void* const* d_in, const int* in_sizes, int n_in,
                              void* d_out, int out_size, void* d_ws, size_t ws_size,
                              hipStream_t stream) {
  (void)in_sizes; (void)n_in; (void)out_size; (void)ws_size;
  const float* src   = (const float*)d_in[0];
  const float* Wq    = (const float*)d_in[1];
  const float* Wk    = (const float*)d_in[2];
  const float* Wv    = (const float*)d_in[3];
  const float* bq    = (const float*)d_in[4];
  const float* bk    = (const float*)d_in[5];
  const float* bv    = (const float*)d_in[6];
  const float* Wo    = (const float*)d_in[7];
  const float* bo    = (const float*)d_in[8];
  const float* rpe   = (const float*)d_in[9];
  const float* W1    = (const float*)d_in[10];
  const float* b1    = (const float*)d_in[11];
  const float* W2    = (const float*)d_in[12];
  const float* b2    = (const float*)d_in[13];
  const float* ln1_g = (const float*)d_in[14];
  const float* ln1_b = (const float*)d_in[15];
  const float* ln2_g = (const float*)d_in[16];
  const float* ln2_b = (const float*)d_in[17];

  char* w = (char*)d_ws;
  const size_t MB = (size_t)1 << 20;
  bf16_t* w1T  = (bf16_t*)(w + 0 * MB);    // [DFF][D]     8 MB
  bf16_t* w2T  = (bf16_t*)(w + 8 * MB);    // [D][DFF]     8 MB
  bf16_t* srcB = (bf16_t*)(w + 16 * MB);   // 8 MB
  bf16_t* wqT  = (bf16_t*)(w + 24 * MB);   // 2 MB
  bf16_t* wkT  = (bf16_t*)(w + 26 * MB);   // 2 MB
  bf16_t* wvT  = (bf16_t*)(w + 28 * MB);   // 2 MB
  bf16_t* qB   = (bf16_t*)(w + 30 * MB);   // 8 MB
  bf16_t* kBp  = (bf16_t*)(w + 38 * MB);   // 8 MB
  bf16_t* vBp  = (bf16_t*)(w + 46 * MB);   // 8 MB (transposed [B,H,Dh,SEQ])
  bf16_t* hB   = (bf16_t*)(w + 16 * MB);   // 32 MB, aliases pool above
  bf16_t* woT  = (bf16_t*)(w + 54 * MB);   // 2 MB
  bf16_t* ctxB = (bf16_t*)(w + 56 * MB);   // 8 MB
  float*  attnF = (float*)(w + 64 * MB);   // 16 MB
  float*  ffF   = (float*)(w + 64 * MB);   // aliases attnF
  float*  xF    = (float*)(w + 80 * MB);   // 16 MB
  bf16_t* xB    = (bf16_t*)(w + 96 * MB);  // 8 MB

  k_cvt<<<(ROWS * D_MODEL + 255) / 256, 256, 0, stream>>>(src, srcB, ROWS * D_MODEL);
  auto cvtT = [&](const float* s, bf16_t* d, int Kd, int Nd) {
    k_cvt_t<<<dim3(Nd / 32, Kd / 32), 256, 0, stream>>>(s, d, Kd, Nd);
  };
  cvtT(Wq, wqT, D_MODEL, D_MODEL);
  cvtT(Wk, wkT, D_MODEL, D_MODEL);
  cvtT(Wv, wvT, D_MODEL, D_MODEL);
  cvtT(Wo, woT, D_MODEL, D_MODEL);
  cvtT(W1, w1T, D_MODEL, DFF);     // -> [DFF][D_MODEL]
  cvtT(W2, w2T, DFF, D_MODEL);     // -> [D_MODEL][DFF]

  const dim3 blk(256);
  const dim3 gD(D_MODEL / TN, ROWS / TM);  // (8, 32)
  const dim3 gF(DFF / TN, ROWS / TM);      // (32, 32)

  k_gemm<2><<<gD, blk, 0, stream>>>(srcB, wqT, bq, nullptr, qB, ROWS, D_MODEL, D_MODEL);
  k_gemm<2><<<gD, blk, 0, stream>>>(srcB, wkT, bk, nullptr, kBp, ROWS, D_MODEL, D_MODEL);
  k_gemm<3><<<gD, blk, 0, stream>>>(srcB, wvT, bv, nullptr, vBp, ROWS, D_MODEL, D_MODEL);

  k_attn<<<dim3(SEQ / 64, BATCH * NHEAD), 128, 0, stream>>>(qB, kBp, vBp, rpe, ctxB);

  k_gemm<0><<<gD, blk, 0, stream>>>(ctxB, woT, bo, attnF, nullptr, ROWS, D_MODEL, D_MODEL);
  k_add_ln<<<ROWS, 256, 0, stream>>>(src, attnF, ln1_g, ln1_b, xF, xB);
  k_gemm<1><<<gF, blk, 0, stream>>>(xB, w1T, b1, nullptr, hB, ROWS, DFF, D_MODEL);
  k_gemm<0><<<gD, blk, 0, stream>>>(hB, w2T, b2, ffF, nullptr, ROWS, D_MODEL, DFF);
  k_add_ln<<<ROWS, 256, 0, stream>>>(xF, ffF, ln2_g, ln2_b, (float*)d_out, nullptr);
}
